// Attention_46514495816386
// MI455X (gfx1250) — compile-verified, matrix operand order checked by reading, not run on hardware
//
#include <hip/hip_runtime.h>

typedef __bf16 bf16_t;
typedef __attribute__((ext_vector_type(16))) __bf16 bf16x16;
typedef __attribute__((ext_vector_type(8)))  __bf16 bf16x8;
typedef __attribute__((ext_vector_type(8)))  float  v8f;

#define C_DIM 512
#define N_TOK 4096
#define B_SZ  4
#define NG    32
#define CPG   16          // channels per group
#define KT    16          // C / 32 (number of K=32 chunks per dot product)

// Load a 16x32 bf16 A/B fragment from a row-major matrix (global or LDS).
// 'rowk' points at (this lane's row)*ld + k0. ISA layout (16-bit A 16x32):
//   lanes 0-15 : VGPR0-3 = K 0..7,  VGPR4-7 = K 16..23
//   lanes 16-31: VGPR0-3 = K 8..15, VGPR4-7 = K 24..31
// so each lane does two 16-byte loads at +kc and +16+kc, kc = (lane/16)*8.
__device__ __forceinline__ bf16x16 ld_frag(const bf16_t* rowk, int kc) {
  bf16x8 lo = *(const bf16x8*)(rowk + kc);
  bf16x8 hi = *(const bf16x8*)(rowk + 16 + kc);
  bf16x16 f;
#pragma unroll
  for (int i = 0; i < 8; ++i) { f[i] = lo[i]; f[i + 8] = hi[i]; }
  return f;
}

__device__ __forceinline__ v8f wmma_bf16(bf16x16 a, bf16x16 b, v8f c) {
  return __builtin_amdgcn_wmma_f32_16x16x32_bf16(false, a, false, b, (short)0, c,
                                                 false, false);
}

// Async global->LDS copy, 16B per lane, tracked by ASYNCcnt.
// vdst = VGPR with LDS byte address, vaddr = 64-bit global address, saddr=off.
__device__ __forceinline__ void async_copy_b128(unsigned lds_off, const bf16_t* gsrc) {
  unsigned long long ga = (unsigned long long)(uintptr_t)gsrc;
  asm volatile("global_load_async_to_lds_b128 %0, %1, off"
               :: "v"(lds_off), "v"(ga) : "memory");
}

// ---------------- Kernel 1: weights fp32 -> bf16 ----------------
__global__ void cvt_w_kernel(const float* __restrict__ wq, const float* __restrict__ wk,
                             const float* __restrict__ wv, const float* __restrict__ wo,
                             bf16_t* wqb, bf16_t* wkb, bf16_t* wvb, bf16_t* wob) {
  int i = blockIdx.x * 256 + threadIdx.x;
  if (i < C_DIM * C_DIM) {
    wqb[i] = (bf16_t)wq[i];
    wkb[i] = (bf16_t)wk[i];
    wvb[i] = (bf16_t)wv[i];
    wob[i] = (bf16_t)wo[i];
  }
}

// ---------------- Kernel 2: GroupNorm -> token-major bf16 hf[B,N,C] ----------------
__global__ void gn_to_bf16_kernel(const float* __restrict__ x,
                                  const float* __restrict__ scale,
                                  const float* __restrict__ bias,
                                  bf16_t* __restrict__ hf) {
  const int bg = blockIdx.x;
  const int b = bg / NG, g = bg % NG;
  const float* xp = x + (size_t)(b * C_DIM + g * CPG) * N_TOK;
  const int tid = threadIdx.x;
  const int M = CPG * N_TOK;
  float s = 0.f, ss = 0.f;
  for (int i = tid; i < M; i += 256) { float v = xp[i]; s += v; ss += v * v; }
  __shared__ float rs[256], rq[256];
  rs[tid] = s; rq[tid] = ss;
  __syncthreads();
  for (int off = 128; off > 0; off >>= 1) {
    if (tid < off) { rs[tid] += rs[tid + off]; rq[tid] += rq[tid + off]; }
    __syncthreads();
  }
  const float inv  = 1.0f / (float)M;
  const float mean = rs[0] * inv;
  const float var  = rq[0] * inv - mean * mean;
  const float rstd = rsqrtf(var + 1e-5f);
  for (int i = tid; i < M; i += 256) {
    int c = g * CPG + (i >> 12);         // i / 4096
    int n = i & (N_TOK - 1);
    float v = (xp[i] - mean) * rstd * scale[c] + bias[c];
    hf[((size_t)(b * N_TOK) + n) * C_DIM + c] = (bf16_t)v;
  }
}

// ---------------- Kernel 3: q,k [B,N,C] and vT [B,C,N] projections ----------------
__global__ void __launch_bounds__(256)
qkv_kernel(const bf16_t* __restrict__ hf,
           const bf16_t* __restrict__ wqb, const bf16_t* __restrict__ wkb,
           const bf16_t* __restrict__ wvb,
           const float* __restrict__ bq, const float* __restrict__ bk,
           const float* __restrict__ bv,
           bf16_t* __restrict__ q, bf16_t* __restrict__ k, bf16_t* __restrict__ vT) {
  const int wid  = blockIdx.x * 8 + (threadIdx.x >> 5);
  const int lane = threadIdx.x & 31;
  const int ct   = wid & 31;           // output-channel tile
  const int nt   = (wid >> 5) & 255;   // token tile
  const int b    = wid >> 13;          // batch
  const int row = lane & 15, half = lane >> 4, kc = half * 8;

  const bf16_t* arow = hf + ((size_t)(b * N_TOK) + nt * 16 + row) * C_DIM;
  bf16x16 A[KT];
#pragma unroll
  for (int kk = 0; kk < KT; ++kk) A[kk] = ld_frag(arow + kk * 32, kc);

  const int d = ct * 16 + row;         // this lane's output channel (N dim of tile)
  const bf16_t* W[3]  = { wqb + (size_t)d * C_DIM, wkb + (size_t)d * C_DIM,
                          wvb + (size_t)d * C_DIM };
  const float  bia[3] = { bq[d], bk[d], bv[d] };

#pragma unroll
  for (int m = 0; m < 3; ++m) {
    v8f acc = {};
#pragma unroll
    for (int kk = 0; kk < KT; ++kk) {
      bf16x16 Bf = ld_frag(W[m] + kk * 32, kc);
      acc = wmma_bf16(A[kk], Bf, acc);
    }
    if (m < 2) {
      bf16_t* dst = (m == 0) ? q : k;
#pragma unroll
      for (int r = 0; r < 8; ++r) {
        int n = nt * 16 + half * 8 + r;      // D layout: M = r + 8*half
        dst[((size_t)(b * N_TOK) + n) * C_DIM + d] = (bf16_t)(acc[r] + bia[m]);
      }
    } else {                                  // v stored transposed: vT[b, c, n]
      bf16x8 pk;
#pragma unroll
      for (int r = 0; r < 8; ++r) pk[r] = (bf16_t)(acc[r] + bia[2]);
      const size_t base = ((size_t)(b * C_DIM) + d) * N_TOK + nt * 16 + half * 8;
      *(bf16x8*)(vT + base) = pk;
    }
  }
}

// ---------------- Kernel 4: flash attention ----------------
// Block = 4 waves, 4 consecutive 16-query tiles of one batch. K/V tiles for each
// 32-key step are staged in LDS (double-buffered) with async b128 copies and
// shared by all 4 waves (4x less L2 traffic than private streaming).
// LDS element map (bf16): buf i (i=0,1) at i*32768: K tile [32][512] then
// V tile [512][32]; P relayout buffers at 65536 + wave*512.
__global__ void __launch_bounds__(128)
flash_kernel(const bf16_t* __restrict__ q, const bf16_t* __restrict__ k,
             const bf16_t* __restrict__ vT, bf16_t* __restrict__ attn) {
  extern __shared__ bf16_t smem[];
  const int tid  = threadIdx.x;
  const int widx = tid >> 5;
  const int b    = blockIdx.x >> 6;              // 64 blocks per batch
  const int qt   = (blockIdx.x & 63) * 4 + widx; // this wave's query tile
  const int lane = tid & 31;
  const int row = lane & 15, half = lane >> 4, kc = half * 8;

  const unsigned lds_base = (unsigned)(uintptr_t)smem;  // flat addr[31:0] == LDS byte addr
  bf16_t* P = smem + 65536 + widx * 512;

  const bf16_t* qrow = q + ((size_t)(b * N_TOK) + qt * 16 + row) * C_DIM;
  bf16x16 A[KT];
#pragma unroll
  for (int kk = 0; kk < KT; ++kk) A[kk] = ld_frag(qrow + kk * 32, kc);

  v8f acc[32];                          // O accumulator: 32 channel tiles of 16
#pragma unroll
  for (int ct = 0; ct < 32; ++ct) acc[ct] = (v8f){};
  float mrow[8], lrow[8];
#pragma unroll
  for (int r = 0; r < 8; ++r) { mrow[r] = -1e30f; lrow[r] = 0.f; }

  const float scl = 0.044194173824159216f;   // 1/sqrt(512)
  const int NB = N_TOK / 32;

  // ---- async stage of one 32-key tile into buffer `buf` ----
  auto issue_tile = [&](int s0, int buf) {
    const unsigned ldsK = lds_base + (unsigned)buf * 65536u;
    const unsigned ldsV = ldsK + 32768u;
    const bf16_t* kg = k + ((size_t)(b * N_TOK) + s0) * C_DIM;  // contiguous 32KB
#pragma unroll
    for (int j = 0; j < 16; ++j) {
      const int ck = tid + 128 * j;                 // 16B-chunk index, 0..2047
      async_copy_b128(ldsK + (unsigned)ck * 16u, kg + ck * 8);
      const int c = ck >> 2, w = ck & 3;            // V: [c][32] runs of 64B
      const bf16_t* vg = vT + ((size_t)(b * C_DIM) + c) * N_TOK + s0 + w * 8;
      async_copy_b128(ldsV + (unsigned)ck * 16u, vg);
    }
  };

  issue_tile(0, 0);                     // prologue: tile 0 -> buffer 0

  for (int it = 0; it < NB; ++it) {
    const int cur = it & 1;
    if (it + 1 < NB) {
      __syncthreads();                  // all waves done reading buf[cur^1]
      issue_tile((it + 1) * 32, cur ^ 1);
      // 32 newer loads outstanding per wave -> <=32 means tile `it` landed
      asm volatile("s_wait_asynccnt 0x20" ::: "memory");
    } else {
      __syncthreads();
      asm volatile("s_wait_asynccnt 0x0" ::: "memory");
    }
    __syncthreads();                    // buf[cur] visible to all waves

    const bf16_t* Kb = smem + cur * 32768;            // [32][512]
    const bf16_t* Vb = smem + cur * 32768 + 16384;    // [512][32]
    const bf16_t* kr0 = Kb + row * C_DIM;             // S0: s = row
    const bf16_t* kr1 = Kb + (16 + row) * C_DIM;      // S1: s = 16 + row

    v8f S0 = {}, S1 = {};
#pragma unroll
    for (int kk = 0; kk < KT; ++kk) {
      bf16x16 B0 = ld_frag(kr0 + kk * 32, kc);
      bf16x16 B1 = ld_frag(kr1 + kk * 32, kc);
      S0 = wmma_bf16(A[kk], B0, S0);
      S1 = wmma_bf16(A[kk], B1, S1);
    }
    float alpha[8];
#pragma unroll
    for (int r = 0; r < 8; ++r) {
      float a0 = S0[r] * scl, a1 = S1[r] * scl;
      float mx = fmaxf(a0, a1);
#pragma unroll
      for (int msk = 1; msk < 16; msk <<= 1) mx = fmaxf(mx, __shfl_xor(mx, msk, 32));
      float mn = fmaxf(mrow[r], mx);
      alpha[r] = __expf(mrow[r] - mn);
      mrow[r] = mn;
      float p0 = __expf(a0 - mn), p1 = __expf(a1 - mn);
      float rsum = p0 + p1;
#pragma unroll
      for (int msk = 1; msk < 16; msk <<= 1) rsum += __shfl_xor(rsum, msk, 32);
      lrow[r] = lrow[r] * alpha[r] + rsum;
      const int t = r + 8 * half;        // C-layout row this lane owns
      P[t * 32 + row]      = (bf16_t)p0; // s = row
      P[t * 32 + 16 + row] = (bf16_t)p1; // s = 16 + row
    }
    // Reload P as an A-fragment (same-wave DS ops are in-order).
    bf16x8 plo = *(const bf16x8*)(P + row * 32 + kc);
    bf16x8 phi = *(const bf16x8*)(P + row * 32 + 16 + kc);
    bf16x16 Pf;
#pragma unroll
    for (int i = 0; i < 8; ++i) { Pf[i] = plo[i]; Pf[i + 8] = phi[i]; }
    // O = diag(alpha)*O + P @ V
#pragma unroll
    for (int ct = 0; ct < 32; ++ct) {
      bf16x16 Vf = ld_frag(Vb + (ct * 16 + row) * 32, kc);
#pragma unroll
      for (int r = 0; r < 8; ++r) acc[ct][r] *= alpha[r];
      acc[ct] = wmma_bf16(Pf, Vf, acc[ct]);
    }
  }

  float invl[8];
#pragma unroll
  for (int r = 0; r < 8; ++r) invl[r] = 1.0f / lrow[r];
#pragma unroll
  for (int ct = 0; ct < 32; ++ct) {
    const int c = ct * 16 + row;
#pragma unroll
    for (int r = 0; r < 8; ++r) {
      const int t = qt * 16 + half * 8 + r;
      attn[((size_t)(b * N_TOK) + t) * C_DIM + c] = (bf16_t)(acc[ct][r] * invl[r]);
    }
  }
}

// ---------------- Kernel 5: out projection + bias + residual, fp32 out ----------------
__global__ void __launch_bounds__(256)
oproj_kernel(const bf16_t* __restrict__ attn, const bf16_t* __restrict__ wob,
             const float* __restrict__ bo, const float* __restrict__ x,
             float* __restrict__ out) {
  const int wid  = blockIdx.x * 8 + (threadIdx.x >> 5);
  const int lane = threadIdx.x & 31;
  const int ct = wid & 31;
  const int nt = (wid >> 5) & 255;
  const int b  = wid >> 13;
  const int row = lane & 15, half = lane >> 4, kc = half * 8;

  const bf16_t* arow = attn + ((size_t)(b * N_TOK) + nt * 16 + row) * C_DIM;
  const int d = ct * 16 + row;
  const bf16_t* wrow = wob + (size_t)d * C_DIM;
  v8f acc = {};
#pragma unroll
  for (int kk = 0; kk < KT; ++kk) {
    bf16x16 Af = ld_frag(arow + kk * 32, kc);
    bf16x16 Bf = ld_frag(wrow + kk * 32, kc);
    acc = wmma_bf16(Af, Bf, acc);
  }
  const float bd = bo[d];
  const size_t base = ((size_t)(b * C_DIM) + d) * N_TOK + nt * 16 + half * 8;
#pragma unroll
  for (int r = 0; r < 8; ++r)
    out[base + r] = acc[r] + bd + x[base + r];
}

extern "C" void kernel_launch(void* const* d_in, const int* in_sizes, int n_in,
                              void* d_out, int out_size, void* d_ws, size_t ws_size,
                              hipStream_t stream) {
  (void)in_sizes; (void)n_in; (void)out_size; (void)ws_size;
  const float* x   = (const float*)d_in[0];
  const float* gns = (const float*)d_in[1];
  const float* gnb = (const float*)d_in[2];
  const float* wq  = (const float*)d_in[3];
  const float* bq  = (const float*)d_in[4];
  const float* wk  = (const float*)d_in[5];
  const float* bk  = (const float*)d_in[6];
  const float* wv  = (const float*)d_in[7];
  const float* bv  = (const float*)d_in[8];
  const float* wo  = (const float*)d_in[9];
  const float* bo  = (const float*)d_in[10];
  float* out = (float*)d_out;

  char* ws = (char*)d_ws;
  const size_t act = (size_t)B_SZ * N_TOK * C_DIM * sizeof(bf16_t);  // 16 MB
  const size_t wsz = (size_t)C_DIM * C_DIM * sizeof(bf16_t);         // 512 KB
  bf16_t* hf  = (bf16_t*)(ws);
  bf16_t* qb  = (bf16_t*)(ws + 1 * act);
  bf16_t* kb  = (bf16_t*)(ws + 2 * act);
  bf16_t* vTb = (bf16_t*)(ws + 3 * act);
  bf16_t* atb = (bf16_t*)(ws + 4 * act);
  bf16_t* wqb = (bf16_t*)(ws + 5 * act);
  bf16_t* wkb = (bf16_t*)(ws + 5 * act + 1 * wsz);
  bf16_t* wvb = (bf16_t*)(ws + 5 * act + 2 * wsz);
  bf16_t* wob = (bf16_t*)(ws + 5 * act + 3 * wsz);

  cvt_w_kernel<<<(C_DIM * C_DIM + 255) / 256, 256, 0, stream>>>(
      wq, wk, wv, wo, wqb, wkb, wvb, wob);
  gn_to_bf16_kernel<<<B_SZ * NG, 256, 0, stream>>>(x, gns, gnb, hf);
  qkv_kernel<<<B_SZ * (N_TOK / 16) * (C_DIM / 16) / 8, 256, 0, stream>>>(
      hf, wqb, wkb, wvb, bq, bk, bv, qb, kb, vTb);
  // dynamic LDS: 2 x (32KB K + 32KB V) + 4 x 1KB P = 135168 B (< 320KB/WGP)
  flash_kernel<<<B_SZ * (N_TOK / 16) / 4, 128, 135168, stream>>>(qb, kb, vTb, atb);
  oproj_kernel<<<B_SZ * (N_TOK / 16) * (C_DIM / 16) / 8, 256, 0, stream>>>(
      atb, wob, bo, x, out);
}
